// Model_76304388981153
// MI455X (gfx1250) — compile-verified, matrix operand order checked by reading, not run on hardware
//
#include <hip/hip_runtime.h>
#include <hip/hip_bf16.h>
#include <math.h>
#include <stdint.h>

// ---------------- problem constants ----------------
#define BB   64
#define TT   512
#define DD   512
#define HH   256
#define GG   1024   // 4H
#define NH2  512    // 2H
#define UU   9
#define MTOT (BB*TT)   // 32768 rows

typedef __bf16 bf16_t;
typedef __attribute__((ext_vector_type(16))) __bf16 v16bf;
typedef __attribute__((ext_vector_type(8)))  __bf16 v8bf;
typedef __attribute__((ext_vector_type(8)))  float  v8f;

__device__ inline v8f wmma_bf16(v16bf a, v16bf b, v8f c) {
  // D = A(16x32 bf16) x B(32x16 bf16) + C(16x16 f32)
  return __builtin_amdgcn_wmma_f32_16x16x32_bf16(
      /*neg_a=*/false, a, /*neg_b=*/false, b,
      /*c_mod=*/(short)0, c, /*reuse_a=*/false, /*reuse_b=*/false);
}

__device__ inline v16bf combine16(v8bf lo, v8bf hi) {
  return __builtin_shufflevector(lo, hi, 0,1,2,3,4,5,6,7,8,9,10,11,12,13,14,15);
}

// A fragment (16x32 bf16 tile) from row-major bf16, ld = row stride (elems).
// ISA 16-bit A layout: lane<16 -> K {0..7,16..23}; lane>=16 -> K {8..15,24..31}
__device__ inline v16bf load_a_frag(const bf16_t* __restrict__ tile, int ld, int lane) {
  const bf16_t* p = tile + (size_t)(lane & 15) * ld + ((lane >> 4) << 3);
  v8bf lo = *(const v8bf*)(p);
  v8bf hi = *(const v8bf*)(p + 16);
  return combine16(lo, hi);
}

// gfx1250 async copy: 16B global -> LDS, tracked by ASYNCcnt
__device__ inline void async_b128_to_lds(uint32_t lds_addr, const void* gptr) {
  asm volatile("global_load_async_to_lds_b128 %0, %1, off"
               :: "v"(lds_addr), "v"((unsigned long long)(uintptr_t)gptr)
               : "memory");
}
__device__ inline void wait_async_le1() { asm volatile("s_wait_asynccnt 0x1" ::: "memory"); }
__device__ inline void wait_async_le0() { asm volatile("s_wait_asynccnt 0x0" ::: "memory"); }

__device__ inline float sigm(float x) { return 1.f / (1.f + __expf(-x)); }

// ---------------- weight repack: f32 [K,N] -> bf16 B-fragment layout ----------------
// packed index = ((nt*(K/32)+kt)*32 + lane)*16 + e
// lane: N col = lane&15 ; K local = (e<8?e:e+8) + ((lane>=16)?8:0)
__global__ void pack_b_kernel(const float* __restrict__ W, bf16_t* __restrict__ P,
                              int K, int N) {
  int idx = blockIdx.x * blockDim.x + threadIdx.x;
  int total = K * N;
  if (idx >= total) return;
  int e    = idx & 15;
  int lane = (idx >> 4) & 31;
  int tile = idx >> 9;
  int ktn  = K >> 5;
  int nt   = tile / ktn;
  int kt   = tile % ktn;
  int kl   = (e < 8 ? e : e + 8) + ((lane >> 4) << 3);
  int k    = kt * 32 + kl;
  int n    = nt * 16 + (lane & 15);
  P[idx] = (bf16_t)W[(size_t)k * N + n];
}

// ---------------- f32 -> bf16 elementwise ----------------
__global__ void cvt_bf16_kernel(const float* __restrict__ src, bf16_t* __restrict__ dst,
                                int n) {
  int i = blockIdx.x * blockDim.x + threadIdx.x;
  if (i < n) dst[i] = (bf16_t)src[i];
}

// ---------------- bf16 WMMA GEMM, 128x64 macro-tile per block --------------------
// 8 waves: wave w owns rows [w*16, w*16+16) of the macro tile and all 4 N-tiles.
// B tiles staged once per block per k-step via async global->LDS, double buffered.
__global__ __launch_bounds__(256) void gemm_bf16_kernel(
    const bf16_t* __restrict__ A,    // row-major [M,K]
    const bf16_t* __restrict__ Bp,   // packed B fragments
    const float*  __restrict__ bias, // [N] or nullptr
    float* __restrict__ C,           // row-major [M,N] f32
    int M, int N, int K, int do_tanh) {
  __shared__ bf16_t Bs[2][4 * 512];  // 2 x 4KB: 4 B-tiles (32 lanes x 16 bf16 each)

  const int tid  = threadIdx.x;
  const int lane = tid & 31;
  const int wv   = tid >> 5;
  const int ktn  = K >> 5;
  const int nBlk = N >> 6;                 // # of 64-wide N macro tiles
  const int bm   = blockIdx.x / nBlk;
  const int bn   = blockIdx.x % nBlk;
  const int m0   = bm * 128 + wv * 16;
  const int nt0  = bn * 4;

  // per-thread staging slot: tile j (0..3), 16B chunk ch (0..63)
  const int j  = tid >> 6;
  const int ch = tid & 63;
  const uint32_t dst0 = (uint32_t)(uintptr_t)&Bs[0][j * 512 + ch * 8];
  const uint32_t dst1 = (uint32_t)(uintptr_t)&Bs[1][j * 512 + ch * 8];
  const bf16_t* srcBase = Bp + ((size_t)(nt0 + j) * ktn) * 512 + ch * 8;

  v8f zero = {};
  v8f acc[4] = {zero, zero, zero, zero};
  const bf16_t* Arow = A + (size_t)m0 * K;

  async_b128_to_lds(dst0, srcBase);                       // stage kt = 0
  for (int kt = 0; kt < ktn; ++kt) {
    const int buf = kt & 1;
    if (kt + 1 < ktn) {                                   // stage kt+1 into other buffer
      async_b128_to_lds(buf ? dst0 : dst1, srcBase + (size_t)(kt + 1) * 512);
      wait_async_le1();                                   // our kt-load has landed
    } else {
      wait_async_le0();
    }
    __syncthreads();                                      // all waves' staging visible

    v16bf a = load_a_frag(Arow + kt * 32, K, lane);
#pragma unroll
    for (int n = 0; n < 4; ++n) {
      v16bf b = *(const v16bf*)(&Bs[buf][n * 512 + lane * 16]);
      acc[n] = wmma_bf16(a, b, acc[n]);
    }
    __syncthreads();                                      // reads done before overwrite
  }

#pragma unroll
  for (int n = 0; n < 4; ++n) {
    int col = (nt0 + n) * 16 + (lane & 15);
    float bv = bias ? bias[col] : 0.f;
#pragma unroll
    for (int r = 0; r < 8; ++r) {
      int row = m0 + r + ((lane >> 4) << 3);
      float v = acc[n][r] + bv;
      if (do_tanh) v = tanhf(v);
      C[(size_t)row * N + col] = v;
    }
  }
}

// ---------------- masked LSTM recurrence (both directions), WMMA on h@Wh -------------
// grid: (4 batch tiles, 2 directions), 256 threads (8 waves).
// Wave w owns h-columns [w*32, w*32+32): N tiles {g*16 + 2w + j} for gate g, j in {0,1}.
__global__ __launch_bounds__(256) void lstm_kernel(
    const float* __restrict__ Zf, const float* __restrict__ Zb,   // [B,T,4H] f32
    const bf16_t* __restrict__ WhPf, const bf16_t* __restrict__ WhPb,
    const unsigned char* __restrict__ mask,                       // [B,T] bool
    float* __restrict__ Hf, float* __restrict__ Hb) {             // [B,T,H] f32
  __shared__ bf16_t hshm[16 * 256];   // h tile, row-major bf16 (A-frag friendly)

  const int dir = blockIdx.y;
  const float*  Z   = dir ? Zb   : Zf;
  const bf16_t* WhP = dir ? WhPb : WhPf;
  float*        Ho  = dir ? Hb   : Hf;

  const int lane = threadIdx.x & 31;
  const int w    = threadIdx.x >> 5;
  const int half = lane >> 4;
  const int nl   = lane & 15;
  const int b0   = blockIdx.x * 16;

  for (int i = threadIdx.x; i < 16 * 256; i += 256) hshm[i] = (bf16_t)0.f;
  __syncthreads();

  v8f cst[2]; v8f hst[2];
  cst[0] = {}; cst[1] = {}; hst[0] = {}; hst[1] = {};

  for (int t = 0; t < TT; ++t) {
    const int te = dir ? (TT - 1 - t) : t;

    // accumulators initialized with precomputed input projection z
    v8f acc[4][2];
#pragma unroll
    for (int g = 0; g < 4; ++g)
#pragma unroll
      for (int jj = 0; jj < 2; ++jj) {
        int col = g * 256 + w * 32 + jj * 16 + nl;
#pragma unroll
        for (int r = 0; r < 8; ++r) {
          int row = b0 + r + half * 8;
          acc[g][jj][r] = Z[((size_t)row * TT + te) * GG + col];
        }
      }

    // h @ Wh : K = 256 -> 8 kt steps, 8 WMMAs each
#pragma unroll
    for (int kt = 0; kt < 8; ++kt) {
      const bf16_t* p = &hshm[(size_t)nl * 256 + kt * 32 + half * 8];
      v8bf lo = *(const v8bf*)p;
      v8bf hi = *(const v8bf*)(p + 16);
      v16bf a = combine16(lo, hi);
#pragma unroll
      for (int g = 0; g < 4; ++g)
#pragma unroll
        for (int jj = 0; jj < 2; ++jj) {
          int ntile = g * 16 + 2 * w + jj;
          v16bf b = *(const v16bf*)(WhP + (((size_t)ntile * 8 + kt) * 32 + lane) * 16);
          acc[g][jj] = wmma_bf16(a, b, acc[g][jj]);
        }
    }
    __syncthreads();  // all LDS reads of h done before overwrite

    // gates + state update (Keras order i,f,g,o), masked carry
#pragma unroll
    for (int jj = 0; jj < 2; ++jj) {
#pragma unroll
      for (int r = 0; r < 8; ++r) {
        int rowl = r + half * 8;
        int row  = b0 + rowl;
        float iv = sigm(acc[0][jj][r]);
        float fv = sigm(acc[1][jj][r]);
        float gv = tanhf(acc[2][jj][r]);
        float ov = sigm(acc[3][jj][r]);
        float cn = fv * cst[jj][r] + iv * gv;
        float hn = ov * tanhf(cn);
        bool  m  = mask[(size_t)row * TT + te] != 0;
        float c2 = m ? cn : cst[jj][r];
        float h2 = m ? hn : hst[jj][r];
        cst[jj][r] = c2;
        hst[jj][r] = h2;
        int col = w * 32 + jj * 16 + nl;
        hshm[rowl * 256 + col] = (bf16_t)h2;
        Ho[((size_t)row * TT + te) * HH + col] = h2;
      }
    }
    __syncthreads();  // h tile complete for next step
  }
}

// ---------------- concat(Hf,Hb) + LayerNorm over 2H=512 ----------------
__global__ __launch_bounds__(256) void ln_kernel(
    const float* __restrict__ Hf, const float* __restrict__ Hb,
    const float* __restrict__ gamma, const float* __restrict__ beta,
    float* __restrict__ Y, bf16_t* __restrict__ Ybf) {
  __shared__ float red[256];
  int row = blockIdx.x;       // b*T + t
  int tid = threadIdx.x;
  int k0 = tid, k1 = tid + 256;
  float x0 = Hf[(size_t)row * 256 + k0];
  float x1 = Hb[(size_t)row * 256 + k0];   // k1-256 == k0
  red[tid] = x0 + x1;
  __syncthreads();
  for (int o = 128; o > 0; o >>= 1) { if (tid < o) red[tid] += red[tid + o]; __syncthreads(); }
  float mu = red[0] * (1.f / 512.f);
  __syncthreads();
  float d0 = x0 - mu, d1 = x1 - mu;
  red[tid] = d0 * d0 + d1 * d1;
  __syncthreads();
  for (int o = 128; o > 0; o >>= 1) { if (tid < o) red[tid] += red[tid + o]; __syncthreads(); }
  float rs = rsqrtf(red[0] * (1.f / 512.f) + 1e-5f);
  float y0 = d0 * rs * gamma[k0] + beta[k0];
  float y1 = d1 * rs * gamma[k1] + beta[k1];
  Y[(size_t)row * 512 + k0] = y0;  Ybf[(size_t)row * 512 + k0] = (bf16_t)y0;
  Y[(size_t)row * 512 + k1] = y1;  Ybf[(size_t)row * 512 + k1] = (bf16_t)y1;
}

// ---------------- attention scores: s[row] = dot(u[row,:], v) (one wave / row) -------
__global__ __launch_bounds__(256) void score_kernel(
    const float* __restrict__ U, const float* __restrict__ v, float* __restrict__ S) {
  int lane = threadIdx.x & 31;
  int wv   = threadIdx.x >> 5;
  int row  = blockIdx.x * 8 + wv;
  float s = 0.f;
  for (int i = lane; i < 512; i += 32) s += U[(size_t)row * 512 + i] * v[i];
#pragma unroll
  for (int o = 16; o > 0; o >>= 1) s += __shfl_xor(s, o, 32);
  if (lane == 0) S[row] = s;
}

// ---------------- softmax over T per batch row ----------------
__global__ __launch_bounds__(256) void softmax_kernel(
    const float* __restrict__ S, float* __restrict__ A) {
  __shared__ float red[256];
  int b = blockIdx.x, tid = threadIdx.x;
  float a0 = S[(size_t)b * 512 + tid];
  float a1 = S[(size_t)b * 512 + tid + 256];
  red[tid] = fmaxf(a0, a1);
  __syncthreads();
  for (int o = 128; o > 0; o >>= 1) { if (tid < o) red[tid] = fmaxf(red[tid], red[tid + o]); __syncthreads(); }
  float mx = red[0];
  __syncthreads();
  float e0 = __expf(a0 - mx), e1 = __expf(a1 - mx);
  red[tid] = e0 + e1;
  __syncthreads();
  for (int o = 128; o > 0; o >>= 1) { if (tid < o) red[tid] += red[tid + o]; __syncthreads(); }
  float inv = 1.f / red[0];
  A[(size_t)b * 512 + tid]       = e0 * inv;
  A[(size_t)b * 512 + tid + 256] = e1 * inv;
}

// ---------------- xw = alpha * x ----------------
__global__ void scale_kernel(const float* __restrict__ LN, const float* __restrict__ A,
                             float* __restrict__ X) {
  size_t i = (size_t)blockIdx.x * blockDim.x + threadIdx.x;
  if (i >= (size_t)MTOT * 512) return;
  X[i] = A[i >> 9] * LN[i];
}

// ---------------- final projection: out = xw @ W2 + b2 (N=9, VALU) ----------------
__global__ void final_kernel(const float* __restrict__ XW, const float* __restrict__ W2,
                             const float* __restrict__ b2, float* __restrict__ out) {
  int i = blockIdx.x * blockDim.x + threadIdx.x;
  if (i >= MTOT * UU) return;
  int u = i % UU, row = i / UU;
  float s = b2[u];
  for (int k = 0; k < 512; ++k) s += XW[(size_t)row * 512 + k] * W2[(size_t)k * UU + u];
  out[i] = s;
}

// =====================================================================================
extern "C" void kernel_launch(void* const* d_in, const int* in_sizes, int n_in,
                              void* d_out, int out_size, void* d_ws, size_t ws_size,
                              hipStream_t stream) {
  const float* x    = (const float*)d_in[0];
  const unsigned char* mask = (const unsigned char*)d_in[1];
  const float* Wi_f = (const float*)d_in[2];
  const float* Wh_f = (const float*)d_in[3];
  const float* b_f  = (const float*)d_in[4];
  const float* Wi_b = (const float*)d_in[5];
  const float* Wh_b = (const float*)d_in[6];
  const float* b_b  = (const float*)d_in[7];
  const float* gamma= (const float*)d_in[8];
  const float* beta = (const float*)d_in[9];
  const float* W1   = (const float*)d_in[10];
  const float* b1   = (const float*)d_in[11];
  const float* av   = (const float*)d_in[12];
  const float* W2   = (const float*)d_in[13];
  const float* b2   = (const float*)d_in[14];
  float* out = (float*)d_out;

  char* ws = (char*)d_ws;
  size_t off = 0;
  auto alloc = [&](size_t bytes) { size_t o = off; off += (bytes + 255) & ~(size_t)255; return o; };

  bf16_t* WiFp = (bf16_t*)(ws + alloc((size_t)DD * GG * 2));
  bf16_t* WiBp = (bf16_t*)(ws + alloc((size_t)DD * GG * 2));
  bf16_t* WhFp = (bf16_t*)(ws + alloc((size_t)HH * GG * 2));
  bf16_t* WhBp = (bf16_t*)(ws + alloc((size_t)HH * GG * 2));
  bf16_t* W1p  = (bf16_t*)(ws + alloc((size_t)NH2 * NH2 * 2));
  float*  Xcur = (float*) (ws + alloc((size_t)MTOT * 512 * 4));
  bf16_t* Xbf  = (bf16_t*)(ws + alloc((size_t)MTOT * 512 * 2));
  float*  Zf   = (float*) (ws + alloc((size_t)MTOT * GG * 4));
  float*  Zb   = (float*) (ws + alloc((size_t)MTOT * GG * 4));
  float*  Hf   = (float*) (ws + alloc((size_t)MTOT * HH * 4));
  float*  Hb   = (float*) (ws + alloc((size_t)MTOT * HH * 4));
  float*  LN   = (float*) (ws + alloc((size_t)MTOT * 512 * 4));
  bf16_t* LNbf = (bf16_t*)(ws + alloc((size_t)MTOT * 512 * 2));
  float*  Ubuf = (float*) (ws + alloc((size_t)MTOT * 512 * 4));
  float*  Sc   = (float*) (ws + alloc((size_t)MTOT * 4));
  float*  Al   = (float*) (ws + alloc((size_t)MTOT * 4));
  (void)ws_size; (void)in_sizes; (void)n_in; (void)out_size; (void)av;

  // ---- pack weights into WMMA B-fragment layout (once per launch, deterministic) ----
  pack_b_kernel<<<(DD * GG + 255) / 256, 256, 0, stream>>>(Wi_f, WiFp, DD, GG);
  pack_b_kernel<<<(DD * GG + 255) / 256, 256, 0, stream>>>(Wi_b, WiBp, DD, GG);
  pack_b_kernel<<<(HH * GG + 255) / 256, 256, 0, stream>>>(Wh_f, WhFp, HH, GG);
  pack_b_kernel<<<(HH * GG + 255) / 256, 256, 0, stream>>>(Wh_b, WhBp, HH, GG);
  pack_b_kernel<<<(NH2 * NH2 + 255) / 256, 256, 0, stream>>>(W1, W1p, NH2, NH2);

  const int nElemX = MTOT * 512;
  for (int layer = 0; layer < 3; ++layer) {
    const float* src = (layer == 0) ? x : Xcur;

    // layer input -> bf16 (A operand, row-major)
    cvt_bf16_kernel<<<(nElemX + 255) / 256, 256, 0, stream>>>(src, Xbf, nElemX);

    // input projections: Z = X @ Wi + b   (M=32768, K=512, N=1024)
    {
      int blocks = (MTOT / 128) * (GG / 64);
      gemm_bf16_kernel<<<blocks, 256, 0, stream>>>(Xbf, WiFp, b_f, Zf, MTOT, GG, DD, 0);
      gemm_bf16_kernel<<<blocks, 256, 0, stream>>>(Xbf, WiBp, b_b, Zb, MTOT, GG, DD, 0);
    }

    // recurrence: 4 batch tiles x 2 directions, WMMA on h@Wh each step
    lstm_kernel<<<dim3(4, 2), 256, 0, stream>>>(Zf, Zb, WhFp, WhBp, mask, Hf, Hb);

    // concat + LayerNorm
    ln_kernel<<<MTOT, 256, 0, stream>>>(Hf, Hb, gamma, beta, LN, LNbf);

    // attention: u = tanh(LN @ W1 + b1)   (M=32768, K=512, N=512)
    {
      int blocks = (MTOT / 128) * (NH2 / 64);
      gemm_bf16_kernel<<<blocks, 256, 0, stream>>>(LNbf, W1p, b1, Ubuf, MTOT, NH2, NH2, 1);
    }
    score_kernel<<<MTOT / 8, 256, 0, stream>>>(Ubuf, av, Sc);
    softmax_kernel<<<BB, 256, 0, stream>>>(Sc, Al);

    // xw = alpha * x  -> next layer input (or final projection input)
    scale_kernel<<<(nElemX + 255) / 256, 256, 0, stream>>>(LN, Al, Xcur);

    if (layer == 2) {
      final_kernel<<<(MTOT * UU + 255) / 256, 256, 0, stream>>>(Xcur, W2, b2, out);
    }
  }
}
